// OMENv2_63024350101873
// MI455X (gfx1250) — matrix operand order repaired and checked by reading, not compiled, not run on hardware
//
#include <hip/hip_runtime.h>
#include <hip/hip_bf16.h>

// ---------------------------------------------------------------------------
// MI455X (gfx1250): bf16 WMMA GEMMs (TDM-staged LDS tiles) + flash dual-softmax
// attention. wave32; 16x16x32 bf16 WMMA; no TxT score materialization.
// ---------------------------------------------------------------------------

typedef __bf16 bf16_t;
typedef __bf16 v16bf __attribute__((ext_vector_type(16)));
typedef float  v8f   __attribute__((ext_vector_type(8)));
typedef unsigned int v4u __attribute__((ext_vector_type(4)));

union FragAB {              // 16 bf16 = 32 bytes = 8 VGPRs (A or B operand)
    v16bf v;
    v4u   q[2];
    unsigned short s[16];
};

#define T_ 2048
#define D_ 1024
#define H_ 16
#define DH_ 64
#define DFF_ 4096
#define NEG_ (-1e30f)

__device__ __forceinline__ unsigned short f2bf(float f) {
    union { float f; unsigned int u; } v; v.f = f;
    unsigned int u = v.u;
    unsigned int r = u + 0x7FFFu + ((u >> 16) & 1u);   // round-to-nearest-even
    return (unsigned short)(r >> 16);
}

__device__ __forceinline__ float fast_rcp(float x) {   // v_rcp_f32 (1 ulp)
    return __builtin_amdgcn_rcpf(x);
}
__device__ __forceinline__ float fast_sigmoid(float x) {
    return fast_rcp(1.f + __expf(-x));
}

// ---------------- Tensor Data Mover staging (gfx1250 TDM) -------------------
#if defined(__has_builtin)
#if __has_builtin(__builtin_amdgcn_tensor_load_to_lds) && __has_builtin(__builtin_amdgcn_s_wait_tensorcnt)
#define USE_TDM 1
#endif
#endif
#ifndef USE_TDM
#define USE_TDM 0
#endif

#if USE_TDM
typedef unsigned int u32x4 __attribute__((ext_vector_type(4)));
typedef int          i32x4 __attribute__((ext_vector_type(4)));
typedef int          i32x8 __attribute__((ext_vector_type(8)));

// 2D tile load: rows x 32 bf16 elements starting at gsrc (row stride K elems),
// into LDS at lds_addr with 32B pad after each 64B row -> 96B LDS row stride.
__device__ __forceinline__ void tdm_load_tile(const unsigned short* gsrc,
                                              unsigned int lds_addr,
                                              int rows, int K) {
    unsigned long long ga = (unsigned long long)(size_t)gsrc;
    u32x4 g0;
    g0[0] = 1u;                                              // count=1 (valid), user D#
    g0[1] = lds_addr;                                        // lds_addr[31:0]
    g0[2] = (unsigned int)(ga & 0xFFFFFFFFu);                // global_addr[31:0]
    g0[3] = (unsigned int)((ga >> 32) & 0x1FFFFFFu)          // global_addr[56:32]
          | (2u << 30);                                      // type=2 ("image")
    i32x8 g1;
    g1[0] = (1 << 16)          // data_size = 2 bytes
          | (1 << 20)          // pad_enable
          | (3 << 22)          // pad_interval: every 16 DWORDs (64B row)
          | (7 << 25);         // pad_amount: 8 DWORDs (32B)
    g1[1] = (int)((unsigned)K << 16);                        // tensor_dim0[15:0] = K
    g1[2] = (int)(((unsigned)rows & 0xFFFFu) << 16)          // tensor_dim1[15:0]
          | (int)((unsigned)K >> 16);                        // tensor_dim0[31:16]
    g1[3] = (32 << 16);                                      // tile_dim0 = 32
    g1[4] = rows;                                            // tile_dim1, tile_dim2=0
    g1[5] = K;                                               // tensor_dim0_stride[31:0]
    g1[6] = 0;
    g1[7] = 0;
    i32x4 z4 = (i32x4){0, 0, 0, 0};
#if __clang_major__ >= 23
    i32x8 z8 = (i32x8){0, 0, 0, 0, 0, 0, 0, 0};
    __builtin_amdgcn_tensor_load_to_lds(g0, g1, z4, z4, z8, 0);
#else
    __builtin_amdgcn_tensor_load_to_lds(g0, g1, z4, z4, 0);
#endif
}
#endif

// -------------------------------- cast f32 -> bf16 -------------------------
__global__ __launch_bounds__(256)
void k_cast_bf16(const float* __restrict__ src, unsigned short* __restrict__ dst, size_t n) {
    for (size_t i = (size_t)blockIdx.x * 256 + threadIdx.x; i < n; i += (size_t)gridDim.x * 256)
        dst[i] = f2bf(src[i]);
}

// -------------------------------- LayerNorm --------------------------------
__global__ __launch_bounds__(256)
void k_layernorm(const float* __restrict__ x, const float* __restrict__ g,
                 const float* __restrict__ b, unsigned short* __restrict__ out, int D) {
    __shared__ float red[32];
    const int row = blockIdx.x, tid = threadIdx.x;
    const float* xr = x + (size_t)row * D;
    float s = 0.f, ss = 0.f;
    for (int i = tid; i < D; i += 256) { float v = xr[i]; s += v; ss += v * v; }
    #pragma unroll
    for (int d = 1; d < 32; d <<= 1) { s += __shfl_xor(s, d, 32); ss += __shfl_xor(ss, d, 32); }
    const int wid = tid >> 5, lane = tid & 31;
    if (lane == 0) { red[wid] = s; red[8 + wid] = ss; }
    __syncthreads();
    if (tid == 0) {
        float ts = 0.f, tss = 0.f;
        for (int w = 0; w < 8; ++w) { ts += red[w]; tss += red[8 + w]; }
        red[16] = ts; red[17] = tss;
    }
    __syncthreads();
    const float mu  = red[16] / D;
    const float var = red[17] / D - mu * mu;
    const float inv = rsqrtf(var + 1e-5f);
    for (int i = tid; i < D; i += 256)
        out[(size_t)row * D + i] = f2bf((xr[i] - mu) * inv * g[i] + b[i]);
}

// ---------------- generic bf16 WMMA GEMM: C = act(A @ W^T + bias) (+resid) --
// A: MxK bf16 row-major, W: NxK bf16 row-major. Block tile 128x64, 8 waves,
// each wave 32x32 via 2x2 grid of 16x16x32 bf16 WMMAs per K-step of 32.
// Tiles staged into LDS by the Tensor Data Mover (one wave issues, all sync).
template<bool BIAS, bool GELU, bool RESID, bool OUTF, bool OUTB>
__global__ __launch_bounds__(256)
void k_gemm_bf16(const unsigned short* __restrict__ A,
                 const unsigned short* __restrict__ W,
                 const float* __restrict__ bias,
                 const float* __restrict__ resid,
                 float* __restrict__ outF,
                 unsigned short* __restrict__ outB,
                 int M, int N, int K) {
    __shared__ __align__(16) unsigned short As[128][48];   // 96B stride, b128 aligned
    __shared__ __align__(16) unsigned short Ws[64][48];
    const int tid = threadIdx.x;
    const int bm = blockIdx.x, bn = blockIdx.y;
    const int lane = tid & 31, wid = tid >> 5;
    const int wm = wid & 3, wn = wid >> 2;        // 4x2 wave grid
    const int nl = lane & 15, kh = lane >> 4;     // column-in-tile, lane half

    v8f acc[2][2];
    #pragma unroll
    for (int mi = 0; mi < 2; ++mi)
        #pragma unroll
        for (int ni = 0; ni < 2; ++ni)
            acc[mi][ni] = (v8f){0.f,0.f,0.f,0.f,0.f,0.f,0.f,0.f};

    for (int k0 = 0; k0 < K; k0 += 32) {
#if USE_TDM
        if (tid == 0) {
            tdm_load_tile(A + (size_t)(bm * 128) * K + k0,
                          (unsigned int)(size_t)&As[0][0], 128, K);
            tdm_load_tile(W + (size_t)(bn * 64) * K + k0,
                          (unsigned int)(size_t)&Ws[0][0], 64, K);
            __builtin_amdgcn_s_wait_tensorcnt(0);
        }
        __syncthreads();
#else
        // manual staging: A tile 128x32 (512 b128), W tile 64x32 (256 b128)
        #pragma unroll
        for (int r = 0; r < 2; ++r) {
            int q = tid * 2 + r;
            int row = q >> 2, cq = q & 3;
            *(v4u*)(&As[row][cq * 8]) =
                *(const v4u*)(A + (size_t)(bm * 128 + row) * K + k0 + cq * 8);
        }
        {
            int row = tid >> 2, cq = tid & 3;
            *(v4u*)(&Ws[row][cq * 8]) =
                *(const v4u*)(W + (size_t)(bn * 64 + row) * K + k0 + cq * 8);
        }
        if (k0 + 32 < K) {      // prefetch next K tile (global_prefetch_b8)
            #pragma unroll
            for (int r = 0; r < 2; ++r) {
                int q = tid * 2 + r;
                __builtin_prefetch(A + (size_t)(bm * 128 + (q >> 2)) * K + k0 + 32 + (q & 3) * 8, 0, 3);
            }
            __builtin_prefetch(W + (size_t)(bn * 64 + (tid >> 2)) * K + k0 + 32 + (tid & 3) * 8, 0, 3);
        }
        __syncthreads();
#endif

        FragAB af[2], wf[2];
        #pragma unroll
        for (int mi = 0; mi < 2; ++mi) {            // A frag: ISA 16x32 layout
            int row = wm * 32 + mi * 16 + nl, kb = kh * 8;
            af[mi].q[0] = *(const v4u*)(&As[row][kb]);        // K kb..kb+7
            af[mi].q[1] = *(const v4u*)(&As[row][kb + 16]);   // K kb+16..kb+23
        }
        #pragma unroll
        for (int ni = 0; ni < 2; ++ni) {            // B frag: col=nl, K half=kh
            int row = wn * 32 + ni * 16 + nl;
            wf[ni].q[0] = *(const v4u*)(&Ws[row][kh * 8]);
            wf[ni].q[1] = *(const v4u*)(&Ws[row][kh * 8 + 16]);
        }
        #pragma unroll
        for (int mi = 0; mi < 2; ++mi)
            #pragma unroll
            for (int ni = 0; ni < 2; ++ni)
                acc[mi][ni] = __builtin_amdgcn_wmma_f32_16x16x32_bf16(
                    false, af[mi].v, false, wf[ni].v, (short)0, acc[mi][ni], false, false);
        __syncthreads();
    }

    // epilogue: C layout — lane holds col nl, rows kh*8 + i
    #pragma unroll
    for (int mi = 0; mi < 2; ++mi) {
        #pragma unroll
        for (int ni = 0; ni < 2; ++ni) {
            const int n = bn * 64 + wn * 32 + ni * 16 + nl;
            const float bv = BIAS ? bias[n] : 0.f;
            #pragma unroll
            for (int i = 0; i < 8; ++i) {
                const int m = bm * 128 + wm * 32 + mi * 16 + kh * 8 + i;
                float v = acc[mi][ni][i] + bv;
                if (GELU)  v = 0.5f * v * (1.f + erff(v * 0.70710678118f));
                if (RESID) v += resid[(size_t)m * N + n];
                if (OUTF)  outF[(size_t)m * N + n] = v;
                if (OUTB)  outB[(size_t)m * N + n] = f2bf(v);
            }
        }
    }
}

// ---------------- flash dual-softmax attention ------------------------------
// One wave per (head, 16 query rows). S = Q K^T via WMMA; two online-softmax
// branches (plain, sigmoid(log_mask)-scaled) share S and the V fragments.
__global__ __launch_bounds__(32)
void k_attn(const unsigned short* __restrict__ qkv,      // (T, 3D) bf16
            const float* __restrict__ log_mask,          // (H, T, T) f32
            unsigned short* __restrict__ cat) {          // (T, 2D) bf16
    __shared__ __align__(16) unsigned short Ps[16][48];
    const int qt = blockIdx.x, h = blockIdx.y;
    const int lane = threadIdx.x;
    const int nl = lane & 15, kh = lane >> 4;
    const int t0 = qt * 16;
    const int mbase = kh * 8;

    // Q A-fragments (16 rows x 64 dh, split into two K=32 chunks)
    FragAB qa[2];
    {
        const unsigned short* Qr = qkv + (size_t)(t0 + nl) * (3 * D_) + h * DH_;
        const int kb = kh * 8;
        #pragma unroll
        for (int c = 0; c < 2; ++c) {
            qa[c].q[0] = *(const v4u*)(Qr + 32 * c + kb);
            qa[c].q[1] = *(const v4u*)(Qr + 32 * c + kb + 16);
        }
    }

    v8f o[2][4];
    float mrun[2][8], lrun[2][8];
    #pragma unroll
    for (int br = 0; br < 2; ++br) {
        #pragma unroll
        for (int dt = 0; dt < 4; ++dt) o[br][dt] = (v8f){0.f,0.f,0.f,0.f,0.f,0.f,0.f,0.f};
        #pragma unroll
        for (int i = 0; i < 8; ++i) { mrun[br][i] = NEG_; lrun[br][i] = 0.f; }
    }

    for (int sb = 0; sb <= t0 + 15; sb += 32) {
        // ---- S = Q K^T : two 16x16 tiles (s-columns sb..+15, sb+16..+31)
        v8f s01[2];
        #pragma unroll
        for (int st = 0; st < 2; ++st) {
            FragAB kbf[2];
            const unsigned short* Kr =
                qkv + (size_t)(sb + st * 16 + nl) * (3 * D_) + D_ + h * DH_;
            #pragma unroll
            for (int c = 0; c < 2; ++c) {
                kbf[c].q[0] = *(const v4u*)(Kr + 32 * c + kh * 8);
                kbf[c].q[1] = *(const v4u*)(Kr + 32 * c + kh * 8 + 16);
            }
            v8f acc = (v8f){0.f,0.f,0.f,0.f,0.f,0.f,0.f,0.f};
            acc = __builtin_amdgcn_wmma_f32_16x16x32_bf16(false, qa[0].v, false, kbf[0].v, (short)0, acc, false, false);
            acc = __builtin_amdgcn_wmma_f32_16x16x32_bf16(false, qa[1].v, false, kbf[1].v, (short)0, acc, false, false);
            s01[st] = acc;
        }
        // ---- V B-fragments (32 keys x 64 dh), shared by both branches
        FragAB vb[4];
        #pragma unroll
        for (int dt = 0; dt < 4; ++dt)
            #pragma unroll
            for (int j = 0; j < 16; ++j)
                vb[dt].s[j] = qkv[(size_t)(sb + kh * 8 + (j & 7) + ((j >> 3) << 4)) * (3 * D_)
                                  + 2 * D_ + h * DH_ + dt * 16 + nl];
        // ---- sigmoid(log_mask) for this tile block (branch 1 only)
        float msig[2][8];
        #pragma unroll
        for (int st = 0; st < 2; ++st) {
            const int scol = sb + st * 16 + nl;
            #pragma unroll
            for (int i = 0; i < 8; ++i) {
                const float lm = log_mask[((size_t)h * T_ + (t0 + mbase + i)) * T_ + scol];
                msig[st][i] = fast_sigmoid(lm);
            }
        }

        #pragma unroll
        for (int br = 0; br < 2; ++br) {
            float xs[2][8];
            #pragma unroll
            for (int st = 0; st < 2; ++st) {
                const int scol = sb + st * 16 + nl;
                #pragma unroll
                for (int i = 0; i < 8; ++i) {
                    const int trow = t0 + mbase + i;
                    float v = s01[st][i] * 0.125f;            // 1/sqrt(64)
                    if (br) v *= msig[st][i];
                    xs[st][i] = (scol > trow) ? NEG_ : v;     // causal
                }
            }
            // online softmax (row reductions across 16-lane halves)
            float alpha[8], mnew[8];
            #pragma unroll
            for (int i = 0; i < 8; ++i) {
                float mx = fmaxf(xs[0][i], xs[1][i]);
                #pragma unroll
                for (int d = 1; d < 16; d <<= 1) mx = fmaxf(mx, __shfl_xor(mx, d, 32));
                mnew[i] = fmaxf(mrun[br][i], mx);
                alpha[i] = __expf(mrun[br][i] - mnew[i]);
                mrun[br][i] = mnew[i];
            }
            float p0[8], p1[8];
            #pragma unroll
            for (int i = 0; i < 8; ++i) {
                p0[i] = __expf(xs[0][i] - mnew[i]);
                p1[i] = __expf(xs[1][i] - mnew[i]);
                float rs = p0[i] + p1[i];
                #pragma unroll
                for (int d = 1; d < 16; d <<= 1) rs += __shfl_xor(rs, d, 32);
                lrun[br][i] = lrun[br][i] * alpha[i] + rs;
            }
            #pragma unroll
            for (int dt = 0; dt < 4; ++dt)
                #pragma unroll
                for (int i = 0; i < 8; ++i) o[br][dt][i] *= alpha[i];
            // transpose P (C layout -> A layout) through LDS
            __syncthreads();
            #pragma unroll
            for (int i = 0; i < 8; ++i) {
                Ps[mbase + i][nl]      = f2bf(p0[i]);
                Ps[mbase + i][16 + nl] = f2bf(p1[i]);
            }
            __syncthreads();
            FragAB pa;
            pa.q[0] = *(const v4u*)(&Ps[nl][kh * 8]);
            pa.q[1] = *(const v4u*)(&Ps[nl][kh * 8 + 16]);
            #pragma unroll
            for (int dt = 0; dt < 4; ++dt)
                o[br][dt] = __builtin_amdgcn_wmma_f32_16x16x32_bf16(
                    false, pa.v, false, vb[dt].v, (short)0, o[br][dt], false, false);
        }
    }

    // write O * (1/l) into cat: txt -> cols [0,D), cau -> cols [D,2D)
    #pragma unroll
    for (int br = 0; br < 2; ++br) {
        float invl[8];
        #pragma unroll
        for (int i = 0; i < 8; ++i) invl[i] = fast_rcp(lrun[br][i]);
        #pragma unroll
        for (int dt = 0; dt < 4; ++dt)
            #pragma unroll
            for (int i = 0; i < 8; ++i) {
                const int trow = t0 + mbase + i;
                const int dcol = dt * 16 + nl;
                cat[(size_t)trow * (2 * D_) + br * D_ + h * DH_ + dcol] =
                    f2bf(o[br][dt][i] * invl[i]);
            }
    }
}

// ---------------- sparsity scalar: lambda * mean(sigmoid(log_mask)) ---------
__global__ void k_zero_scalar(float* p) { if (threadIdx.x == 0 && blockIdx.x == 0) p[0] = 0.f; }

__global__ __launch_bounds__(256)
void k_sp_reduce(const float* __restrict__ lm, float* __restrict__ out, size_t n, float scale) {
    __shared__ float red[8];
    float s = 0.f;
    for (size_t i = (size_t)blockIdx.x * 256 + threadIdx.x; i < n; i += (size_t)gridDim.x * 256)
        s += fast_sigmoid(lm[i]);
    #pragma unroll
    for (int d = 1; d < 32; d <<= 1) s += __shfl_xor(s, d, 32);
    if ((threadIdx.x & 31) == 0) red[threadIdx.x >> 5] = s;
    __syncthreads();
    if (threadIdx.x == 0) {
        float t = 0.f;
        for (int w = 0; w < 8; ++w) t += red[w];
        atomicAdd(out, t * scale);
    }
}

// ---------------------------------------------------------------------------
extern "C" void kernel_launch(void* const* d_in, const int* in_sizes, int n_in,
                              void* d_out, int out_size, void* d_ws, size_t ws_size,
                              hipStream_t stream) {
    (void)in_sizes; (void)n_in; (void)out_size; (void)ws_size;
    const float* x      = (const float*)d_in[0];
    const float* W_qkv  = (const float*)d_in[2];
    const float* W_out  = (const float*)d_in[3];
    const float* log_m  = (const float*)d_in[4];
    const float* W_gate = (const float*)d_in[5];
    const float* b_gate = (const float*)d_in[6];
    const float* g1     = (const float*)d_in[7];
    const float* b1     = (const float*)d_in[8];
    const float* g2     = (const float*)d_in[9];
    const float* b2     = (const float*)d_in[10];
    const float* W_ff1  = (const float*)d_in[11];
    const float* b_ff1  = (const float*)d_in[12];
    const float* W_ff2  = (const float*)d_in[13];
    const float* b_ff2  = (const float*)d_in[14];
    float* out = (float*)d_out;

    // workspace carve-up (256-B aligned)
    char* ws = (char*)d_ws;
    size_t off = 0;
    auto carve = [&](size_t bytes) { size_t o = off; off += (bytes + 255) & ~(size_t)255; return o; };
    unsigned short* wqkv = (unsigned short*)(ws + carve((size_t)3 * D_ * D_ * 2));
    unsigned short* wgat = (unsigned short*)(ws + carve((size_t)D_ * 2 * D_ * 2));
    unsigned short* wout = (unsigned short*)(ws + carve((size_t)D_ * D_ * 2));
    unsigned short* wff1 = (unsigned short*)(ws + carve((size_t)DFF_ * D_ * 2));
    unsigned short* wff2 = (unsigned short*)(ws + carve((size_t)D_ * DFF_ * 2));
    unsigned short* xn   = (unsigned short*)(ws + carve((size_t)T_ * D_ * 2));
    unsigned short* qkvb = (unsigned short*)(ws + carve((size_t)T_ * 3 * D_ * 2));
    unsigned short* catb = (unsigned short*)(ws + carve((size_t)T_ * 2 * D_ * 2));
    unsigned short* gb   = (unsigned short*)(ws + carve((size_t)T_ * D_ * 2));
    float*          hb   = (float*)         (ws + carve((size_t)T_ * D_ * 4));
    unsigned short* hnb  = (unsigned short*)(ws + carve((size_t)T_ * D_ * 2));
    unsigned short* ffb  = (unsigned short*)(ws + carve((size_t)T_ * DFF_ * 2));

    // weights -> bf16
    k_cast_bf16<<<1024, 256, 0, stream>>>(W_qkv, wqkv, (size_t)3 * D_ * D_);
    k_cast_bf16<<<1024, 256, 0, stream>>>(W_gate, wgat, (size_t)D_ * 2 * D_);
    k_cast_bf16<<<1024, 256, 0, stream>>>(W_out, wout, (size_t)D_ * D_);
    k_cast_bf16<<<1024, 256, 0, stream>>>(W_ff1, wff1, (size_t)DFF_ * D_);
    k_cast_bf16<<<1024, 256, 0, stream>>>(W_ff2, wff2, (size_t)D_ * DFF_);

    // xn = LN(x) (bf16)
    k_layernorm<<<T_, 256, 0, stream>>>(x, g1, b1, xn, D_);

    // qkv = xn @ W_qkv^T            (2048 x 3072, K=1024)
    k_gemm_bf16<false,false,false,false,true><<<dim3(T_/128, 3*D_/64), 256, 0, stream>>>(
        xn, wqkv, nullptr, nullptr, nullptr, qkvb, T_, 3*D_, D_);

    // dual attention -> cat (2048 x 2048)
    k_attn<<<dim3(T_/16, H_), 32, 0, stream>>>(qkvb, log_m, catb);

    // g = cat @ W_gate^T + b_gate   (2048 x 1024, K=2048)
    k_gemm_bf16<true,false,false,false,true><<<dim3(T_/128, D_/64), 256, 0, stream>>>(
        catb, wgat, b_gate, nullptr, nullptr, gb, T_, D_, 2*D_);

    // h = x + g @ W_out^T           (f32, 2048 x 1024, K=1024)
    k_gemm_bf16<false,false,true,true,false><<<dim3(T_/128, D_/64), 256, 0, stream>>>(
        gb, wout, nullptr, x, hb, nullptr, T_, D_, D_);

    // hn = LN(h)
    k_layernorm<<<T_, 256, 0, stream>>>(hb, g2, b2, hnb, D_);

    // ff = gelu(hn @ W_ff1^T + b_ff1)   (2048 x 4096, K=1024)
    k_gemm_bf16<true,true,false,false,true><<<dim3(T_/128, DFF_/64), 256, 0, stream>>>(
        hnb, wff1, b_ff1, nullptr, nullptr, ffb, T_, DFF_, D_);

    // out = h + ff @ W_ff2^T + b_ff2    (f32 -> d_out, K=4096)
    k_gemm_bf16<true,false,true,true,false><<<dim3(T_/128, D_/64), 256, 0, stream>>>(
        ffb, wff2, b_ff2, hb, out, nullptr, T_, D_, DFF_);

    // sp scalar at d_out[T*D]
    float* sp = out + (size_t)T_ * D_;
    k_zero_scalar<<<1, 32, 0, stream>>>(sp);
    const size_t nlm = (size_t)H_ * T_ * T_;
    k_sp_reduce<<<4096, 256, 0, stream>>>(log_m, sp, nlm, 0.0005f / (float)nlm);
}
